// MultiHeadAttention_74285754352148
// MI455X (gfx1250) — matrix operand order repaired and checked by reading, not compiled, not run on hardware
//
#include <hip/hip_runtime.h>
#include <hip/hip_bf16.h>
#include <stdint.h>

typedef __attribute__((ext_vector_type(16))) __bf16 v16bf;
typedef __attribute__((ext_vector_type(8)))  float  v8f;
typedef __attribute__((ext_vector_type(4)))  unsigned u32x4;
typedef __attribute__((ext_vector_type(8)))  int      i32x8;
typedef __attribute__((ext_vector_type(4)))  int      i32x4;

#define D_MODEL 1024
#define SEQ     2048
#define NHEAD   16
#define HDIM    64
#define WINHALF 256
#define MTOT    4096   // B*S

#if defined(__gfx1250__) && __has_builtin(__builtin_amdgcn_tensor_load_to_lds)
#define USE_TDM 1
#else
#define USE_TDM 0
#endif

struct U8x4 { uint4 lo, hi; };

__device__ inline unsigned short f2bf(float x) {
  unsigned u = __float_as_uint(x);
  u += 0x7FFFu + ((u >> 16) & 1u);          // round-to-nearest-even
  return (unsigned short)(u >> 16);
}

// A-matrix bf16 16x32 fragment (ISA 7.12.2): lane holds row M=lane%16;
// lanes<16: K = koff+{0..7,16..23}, lanes>=16: K = koff+{8..15,24..31}
__device__ inline v16bf ld_frag_a(const unsigned short* base, int stride, int row, int koff) {
  int lane = threadIdx.x & 31;
  const unsigned short* p = base + row * stride + koff + ((lane & 16) ? 8 : 0);
  U8x4 d;
  d.lo = *(const uint4*)(p);
  d.hi = *(const uint4*)(p + 16);
  return __builtin_bit_cast(v16bf, d);
}

// B-matrix bf16 32x16 fragment: lane holds col N=lane%16;
// lanes<16: K = koff+{0..7,16..23}, lanes>=16: K = koff+{8..15,24..31}
__device__ inline v16bf ld_frag_b(const unsigned short* base, int stride, int row, int koff) {
  int lane = threadIdx.x & 31;
  const unsigned short* p = base + row * stride + koff + ((lane & 16) ? 8 : 0);
  U8x4 d;
  d.lo = *(const uint4*)(p);
  d.hi = *(const uint4*)(p + 16);
  return __builtin_bit_cast(v16bf, d);
}

__device__ inline float red_max16(float v) {
  v = fmaxf(v, __shfl_xor(v, 1));
  v = fmaxf(v, __shfl_xor(v, 2));
  v = fmaxf(v, __shfl_xor(v, 4));
  v = fmaxf(v, __shfl_xor(v, 8));
  return v;
}
__device__ inline float red_sum16(float v) {
  v += __shfl_xor(v, 1);
  v += __shfl_xor(v, 2);
  v += __shfl_xor(v, 4);
  v += __shfl_xor(v, 8);
  return v;
}

#if USE_TDM
// TDM: async-DMA a 64x64 bf16 tile (row stride D_MODEL halfwords in memory)
// into LDS with a 4-DWORD pad after each 32-DWORD row -> LDS stride 72 halfs.
__device__ inline void tdm_load_tile64(const void* gsrc, unsigned lds_byte_off) {
  unsigned long long ga = (unsigned long long)(uintptr_t)gsrc;
  // ---- group 0: count=1 | lds_addr | global_addr | type=2 ----
  unsigned long long g0q0 = 1ull | ((unsigned long long)lds_byte_off << 32);
  unsigned long long g0q1 = (ga & 0x1FFFFFFFFFFFFFFull) | (2ull << 62);
  u32x4 g0;
  g0[0] = (unsigned)g0q0; g0[1] = (unsigned)(g0q0 >> 32);
  g0[2] = (unsigned)g0q1; g0[3] = (unsigned)(g0q1 >> 32);
  // ---- group 1 ----
  const unsigned long long dim0 = 1u << 20, dim1 = 1u << 20;   // huge: no OOB clip
  // data_size=2B (1), pad_enable, pad_interval=32 DWORDs (4), pad_amount=4 DWORDs (3)
  unsigned long long q0 = (unsigned long long)((1u << 16) | (1u << 20) | (4u << 22) | (3u << 25))
                        | ((dim0 & 0xFFFFull) << 48);
  unsigned long long q1 = ((dim0 >> 16) & 0xFFFFull) | (dim1 << 16) | (64ull << 48); // tile_dim0=64
  unsigned long long q2 = 64ull /*tile_dim1*/ | ((unsigned long long)(D_MODEL) << 32); // dim0_stride lo
  unsigned long long q3 = 0ull;                                 // dim0_stride hi=0, dim1_stride=0
  i32x8 g1;
  g1[0] = (int)q0; g1[1] = (int)(q0 >> 32);
  g1[2] = (int)q1; g1[3] = (int)(q1 >> 32);
  g1[4] = (int)q2; g1[5] = (int)(q2 >> 32);
  g1[6] = (int)q3; g1[7] = (int)(q3 >> 32);
  i32x4 g2 = {};
  g2[0] = 1 << 20;          // tensor_dim2 large (unused 2D tile, keep non-zero)
  i32x4 g3 = {};
#if __has_include(<hip/amd_detail/amd_gfx1250_TDM.h>)
  i32x8 z8 = {};
  __builtin_amdgcn_tensor_load_to_lds(g0, g1, g2, g3, z8, 0);   // clang-23 6-arg form
#else
  __builtin_amdgcn_tensor_load_to_lds(g0, g1, g2, g3, 0);       // ROCm 7.2 5-arg form
#endif
}
#endif

// AMODE: 0 = fp32 A input, 1 = bf16 A input
// EPI:   0 = bf16 out, 1 = bf16 out + fused RoPE, 2 = fp32 out + bias
template<int AMODE, int EPI>
__global__ __launch_bounds__(256) void gemm_k(const void* Ap, const float* W,
                                              const float* bias, void* Outp) {
  __shared__ __align__(16) unsigned short As[128 * 40];
  __shared__ __align__(16) unsigned short Bs[64 * 40];   // transposed [n][k]
  const int tid   = threadIdx.x;
  const int wid   = tid >> 5;
  const int lane  = tid & 31;
  const int lanem = lane & 15;
  const int laneh = lane >> 4;
  const int m0 = blockIdx.x * 128;
  const int n0 = blockIdx.y * 64;

  v8f acc[4] = {};

  for (int kt = 0; kt < D_MODEL; kt += 32) {
    // ---- stage A tile 128x32 -> bf16 LDS ----
    if (AMODE == 0) {
      const float* A = (const float*)Ap;
      #pragma unroll
      for (int p = 0; p < 4; ++p) {
        int r  = p * 32 + (tid >> 3);
        int kk = (tid & 7) * 4;
        const float* src = A + (size_t)(m0 + r) * D_MODEL + kt + kk;
        float4 f = *(const float4*)src;
        if (kt + 32 < D_MODEL) __builtin_prefetch(src + 32, 0, 1);
        uint2 u;
        u.x = (unsigned)f2bf(f.x) | ((unsigned)f2bf(f.y) << 16);
        u.y = (unsigned)f2bf(f.z) | ((unsigned)f2bf(f.w) << 16);
        *(uint2*)&As[r * 40 + kk] = u;
      }
    } else {
      const unsigned short* A = (const unsigned short*)Ap;
      #pragma unroll
      for (int p = 0; p < 2; ++p) {
        int r  = p * 64 + (tid >> 2);
        int kk = (tid & 3) * 8;
        const unsigned short* src = A + (size_t)(m0 + r) * D_MODEL + kt + kk;
        uint4 v = *(const uint4*)src;
        if (kt + 32 < D_MODEL) __builtin_prefetch(src + 32, 0, 1);
        *(uint4*)&As[r * 40 + kk] = v;
      }
    }
    // ---- stage B tile 32x64 transposed -> Bs[n][k] ----
    #pragma unroll
    for (int p = 0; p < 2; ++p) {
      int kr = p * 16 + (tid >> 4);
      int nc = (tid & 15) * 4;
      const float* src = W + (size_t)(kt + kr) * D_MODEL + n0 + nc;
      float4 f = *(const float4*)src;
      if (kt + 32 < D_MODEL) __builtin_prefetch(src + 32 * D_MODEL, 0, 1);
      Bs[(nc + 0) * 40 + kr] = f2bf(f.x);
      Bs[(nc + 1) * 40 + kr] = f2bf(f.y);
      Bs[(nc + 2) * 40 + kr] = f2bf(f.z);
      Bs[(nc + 3) * 40 + kr] = f2bf(f.w);
    }
    __syncthreads();

    v16bf a = ld_frag_a(As, 40, wid * 16 + lanem, 0);
    #pragma unroll
    for (int t = 0; t < 4; ++t) {
      v16bf b = ld_frag_b(Bs, 40, t * 16 + lanem, 0);
      acc[t] = __builtin_amdgcn_wmma_f32_16x16x32_bf16(false, a, false, b,
                                                       (short)0, acc[t], false, false);
    }
    __syncthreads();
  }

  if (EPI == 1) {
    // fused RoPE: pair (d, d+32) lives in accumulators t and t+2, same lane/reg
    #pragma unroll
    for (int t = 0; t < 2; ++t) {
      #pragma unroll
      for (int r = 0; r < 8; ++r) {
        int d    = t * 16 + lanem;                       // 0..31
        int rowg = m0 + wid * 16 + r + 8 * laneh;
        float pos = (float)(rowg & (SEQ - 1));
        float ang = pos * __powf(10000.0f, -(float)d * (1.0f / 32.0f));
        float sn, cs;
        __sincosf(ang, &sn, &cs);
        float x0 = acc[t][r], x2 = acc[t + 2][r];
        acc[t][r]     = x0 * cs - x2 * sn;
        acc[t + 2][r] = x2 * cs + x0 * sn;
      }
    }
  }

  #pragma unroll
  for (int t = 0; t < 4; ++t) {
    #pragma unroll
    for (int r = 0; r < 8; ++r) {
      int rowg = m0 + wid * 16 + r + 8 * laneh;
      int ncol = n0 + t * 16 + lanem;
      float v = acc[t][r];
      if (EPI == 2)
        ((float*)Outp)[(size_t)rowg * D_MODEL + ncol] = v + bias[ncol];
      else
        ((unsigned short*)Outp)[(size_t)rowg * D_MODEL + ncol] = f2bf(v);
    }
  }
}

// Flash-style sliding-window attention: one (b, h, 64-query block) per WG (4 waves).
__global__ __launch_bounds__(128) void attn_k(const unsigned short* Qb,
                                              const unsigned short* Kb,
                                              const unsigned short* Vb,
                                              unsigned short* Cb) {
  __shared__ __align__(16) unsigned short Qs[64 * 72];
  __shared__ __align__(16) unsigned short Ks[64 * 72];
  __shared__ __align__(16) unsigned short Vs[64 * 72];      // transposed [hd][key]
  __shared__ __align__(16) unsigned short Ps[4][16 * 72];   // per-wave P tile

  const int tid   = threadIdx.x;
  const int wq    = tid >> 5;
  const int lane  = tid & 31;
  const int lanem = lane & 15;
  const int laneh = lane >> 4;
  const int q0 = blockIdx.x * 64;
  const int h  = blockIdx.y;
  const int b  = blockIdx.z;
  const size_t rowbase = (size_t)b * SEQ;

  // ---- stage Q tile (bf16, already RoPE'd) ----
#if USE_TDM
  if (wq == 0)
    tdm_load_tile64(Qb + (rowbase + q0) * D_MODEL + h * HDIM,
                    (unsigned)(uintptr_t)&Qs[0]);
#else
  for (int idx = tid; idx < 64 * 16; idx += 128) {
    int r = idx >> 4, c4 = (idx & 15) * 4;
    *(uint2*)&Qs[r * 72 + c4] =
        *(const uint2*)(Qb + (rowbase + q0 + r) * D_MODEL + h * HDIM + c4);
  }
#endif

  v8f o[4] = {};
  float mrun[8], lrun[8];
  #pragma unroll
  for (int r = 0; r < 8; ++r) { mrun[r] = -1e30f; lrun[r] = 0.0f; }

  int kb0 = q0 - WINHALF; if (kb0 < 0) kb0 = 0;
  for (int kb = kb0; kb <= q0; kb += 64) {
    __syncthreads();                       // protect Ks/Vs reuse
    // ---- stage K tile ----
#if USE_TDM
    if (wq == 0)
      tdm_load_tile64(Kb + (rowbase + kb) * D_MODEL + h * HDIM,
                      (unsigned)(uintptr_t)&Ks[0]);
#else
    for (int idx = tid; idx < 64 * 16; idx += 128) {
      int r = idx >> 4, c4 = (idx & 15) * 4;
      *(uint2*)&Ks[r * 72 + c4] =
          *(const uint2*)(Kb + (rowbase + kb + r) * D_MODEL + h * HDIM + c4);
    }
#endif
    // ---- stage V tile transposed (manual: TDM cannot transpose) ----
    for (int idx = tid; idx < 64 * 16; idx += 128) {
      int r = idx >> 4, c4 = (idx & 15) * 4;
      uint2 raw = *(const uint2*)(Vb + (rowbase + kb + r) * D_MODEL + h * HDIM + c4);
      const unsigned short* t4 = (const unsigned short*)&raw;
      #pragma unroll
      for (int i = 0; i < 4; ++i) Vs[(c4 + i) * 72 + r] = t4[i];
    }
#if USE_TDM
    if (wq == 0) __builtin_amdgcn_s_wait_tensorcnt(0);  // Q (first iter) + K DMA done
#endif
    __syncthreads();

    // scores: S(16x64) = Q_strip(16x64) @ K^T  via 2 k-steps x 4 n-tiles
    v8f sc[4] = {};
    #pragma unroll
    for (int ks = 0; ks < 64; ks += 32) {
      v16bf aq = ld_frag_a(Qs, 72, wq * 16 + lanem, ks);
      #pragma unroll
      for (int t = 0; t < 4; ++t) {
        v16bf bk = ld_frag_b(Ks, 72, t * 16 + lanem, ks);
        sc[t] = __builtin_amdgcn_wmma_f32_16x16x32_bf16(false, aq, false, bk,
                                                        (short)0, sc[t], false, false);
      }
    }
    // scale + causal/window mask (row = r + 8*laneh, col = t*16 + lanem)
    #pragma unroll
    for (int t = 0; t < 4; ++t) {
      #pragma unroll
      for (int r = 0; r < 8; ++r) {
        int rowi = q0 + wq * 16 + r + 8 * laneh;
        int col  = kb + t * 16 + lanem;
        int diff = rowi - col;
        float s = sc[t][r] * 0.125f;       // 1/sqrt(64)
        sc[t][r] = (diff >= 0 && diff <= WINHALF) ? s : -1e30f;
      }
    }
    // online softmax: width-16 lane-group reductions (halves carry different rows)
    #pragma unroll
    for (int r = 0; r < 8; ++r) {
      float mx = fmaxf(fmaxf(sc[0][r], sc[1][r]), fmaxf(sc[2][r], sc[3][r]));
      mx = red_max16(mx);
      float mnew  = fmaxf(mrun[r], mx);
      float alpha = __expf(mrun[r] - mnew);
      float rs = 0.0f;
      #pragma unroll
      for (int t = 0; t < 4; ++t) {
        float p = __expf(sc[t][r] - mnew);
        sc[t][r] = p;
        rs += p;
      }
      rs = red_sum16(rs);
      mrun[r] = mnew;
      lrun[r] = lrun[r] * alpha + rs;
      #pragma unroll
      for (int t = 0; t < 4; ++t) o[t][r] *= alpha;
    }
    // P -> LDS (bf16) so it can be re-read in A-fragment layout
    #pragma unroll
    for (int t = 0; t < 4; ++t)
      #pragma unroll
      for (int r = 0; r < 8; ++r)
        Ps[wq][(r + 8 * laneh) * 72 + t * 16 + lanem] = f2bf(sc[t][r]);
    __syncthreads();

    // O += P(16x64) @ V(64x64): B-fragments from transposed Vs[hd][key]
    #pragma unroll
    for (int ks = 0; ks < 64; ks += 32) {
      v16bf ap = ld_frag_a(Ps[wq], 72, lanem, ks);
      #pragma unroll
      for (int t = 0; t < 4; ++t) {
        v16bf bv = ld_frag_b(Vs, 72, t * 16 + lanem, ks);
        o[t] = __builtin_amdgcn_wmma_f32_16x16x32_bf16(false, ap, false, bv,
                                                       (short)0, o[t], false, false);
      }
    }
  }

  // normalize and store ctx (bf16, [B*S, D] row-major = [B,S,H,HD])
  #pragma unroll
  for (int t = 0; t < 4; ++t) {
    #pragma unroll
    for (int r = 0; r < 8; ++r) {
      float inv = 1.0f / lrun[r];
      int rowg = q0 + wq * 16 + r + 8 * laneh;
      int col  = h * HDIM + t * 16 + lanem;
      Cb[(rowbase + rowg) * D_MODEL + col] = f2bf(o[t][r] * inv);
    }
  }
}

extern "C" void kernel_launch(void* const* d_in, const int* in_sizes, int n_in,
                              void* d_out, int out_size, void* d_ws, size_t ws_size,
                              hipStream_t stream) {
  const float* X  = (const float*)d_in[0];
  const float* Wq = (const float*)d_in[1];
  const float* Wk = (const float*)d_in[2];
  const float* Wv = (const float*)d_in[3];
  const float* Wo = (const float*)d_in[4];
  const float* bo = (const float*)d_in[5];

  unsigned short* Qb = (unsigned short*)d_ws;
  unsigned short* Kb = Qb + (size_t)MTOT * D_MODEL;
  unsigned short* Vb = Kb + (size_t)MTOT * D_MODEL;
  unsigned short* Cb = Vb + (size_t)MTOT * D_MODEL;

  dim3 gg(MTOT / 128, D_MODEL / 64);
  gemm_k<0, 1><<<gg, 256, 0, stream>>>(X, Wq, nullptr, Qb);   // Q + RoPE
  gemm_k<0, 1><<<gg, 256, 0, stream>>>(X, Wk, nullptr, Kb);   // K + RoPE
  gemm_k<0, 0><<<gg, 256, 0, stream>>>(X, Wv, nullptr, Vb);   // V
  attn_k<<<dim3(SEQ / 64, NHEAD, 2), 128, 0, stream>>>(Qb, Kb, Vb, Cb);
  gemm_k<1, 2><<<gg, 256, 0, stream>>>(Cb, Wo, bo, d_out);    // out proj + bias
}
